// BitNetAttention_32358283608478
// MI455X (gfx1250) — compile-verified, matrix operand order checked by reading, not run on hardware
//
#include <hip/hip_runtime.h>
#include <hip/hip_bf16.h>

#define B_   2
#define S_   1024
#define H_   4096
#define NH_  32
#define NKV_ 8
#define HD_  128
#define M_   (B_ * S_)          // 2048
#define NEG_ (-1.0e9f)

typedef __attribute__((ext_vector_type(8)))  __bf16 bhalf8;
typedef __attribute__((ext_vector_type(16))) __bf16 bhalf16;
typedef __attribute__((ext_vector_type(8)))  float  float8;

// ---------------------------------------------------------------------------
// WMMA wrapper: D = A(16x32 bf16) * B(32x16 bf16) + C(16x16 f32)
// ---------------------------------------------------------------------------
__device__ inline float8 wmma_bf16(bhalf16 a, bhalf16 b, float8 c) {
    return __builtin_amdgcn_wmma_f32_16x16x32_bf16(
        /*neg_a=*/false, a, /*neg_b=*/false, b,
        /*c_mod=*/(short)0, c, /*reuse_a=*/false, /*reuse_b=*/false);
}

// ---------------------------------------------------------------------------
// Load a 16x32 bf16 fragment (A layout per ISA 7.12.2; also used for B when
// the B matrix's columns are contiguous in memory, i.e. stored [N][K]).
//   lane 0-15 : row r=lane,     K = k0+{0..7}, k0+{16..23}
//   lane16-31 : row r=lane-16,  K = k0+{8..15}, k0+{24..31}
// ---------------------------------------------------------------------------
__device__ inline bhalf16 load_frag(const __bf16* __restrict__ base, int ld,
                                    int row0, int k0) {
    int lane = threadIdx.x & 31;
    int r    = lane & 15;
    int kh   = (lane & 16) ? 8 : 0;
    const __bf16* p = base + (size_t)(row0 + r) * ld + (k0 + kh);
    bhalf8 lo = *reinterpret_cast<const bhalf8*>(p);
    bhalf8 hi = *reinterpret_cast<const bhalf8*>(p + 16);
    bhalf16 f;
#pragma unroll
    for (int i = 0; i < 8; ++i) { f[i] = lo[i]; f[i + 8] = hi[i]; }
    return f;
}

// ---------------------------------------------------------------------------
// Stage 1: per-block partial abs-sums (deterministic, no atomics)
// ---------------------------------------------------------------------------
__global__ __launch_bounds__(256) void abs_sum_partial(
    const float* __restrict__ w, long long n, float* __restrict__ partials) {
    __shared__ float sm[256];
    float s = 0.f;
    for (long long i = (long long)blockIdx.x * 256 + threadIdx.x; i < n;
         i += 256LL * gridDim.x)
        s += fabsf(w[i]);
    sm[threadIdx.x] = s;
    __syncthreads();
    for (int o = 128; o > 0; o >>= 1) {
        if ((int)threadIdx.x < o) sm[threadIdx.x] += sm[threadIdx.x + o];
        __syncthreads();
    }
    if (threadIdx.x == 0) partials[blockIdx.x] = sm[0];
}

// Stage 2: gamma[w] = mean(|W|) + 1e-5   (one block per weight matrix)
__global__ __launch_bounds__(256) void gamma_finalize(
    const float* __restrict__ partials, float* __restrict__ gammas) {
    __shared__ float sm[256];
    int w = blockIdx.x;
    const long long counts[4] = {(long long)NH_ * HD_ * H_, (long long)NKV_ * HD_ * H_,
                                 (long long)NKV_ * HD_ * H_, (long long)H_ * NH_ * HD_};
    float s = 0.f;
    for (int i = threadIdx.x; i < 512; i += 256) s += partials[w * 512 + i];
    sm[threadIdx.x] = s;
    __syncthreads();
    for (int o = 128; o > 0; o >>= 1) {
        if ((int)threadIdx.x < o) sm[threadIdx.x] += sm[threadIdx.x + o];
        __syncthreads();
    }
    if (threadIdx.x == 0) gammas[w] = sm[0] / (float)counts[w] + 1e-5f;
}

// ---------------------------------------------------------------------------
// Ternary quantize to bf16:  clip(round(w/g), -1, 1) * g   (rint = half-even)
// ---------------------------------------------------------------------------
__global__ __launch_bounds__(256) void quant_bf16(
    const float* __restrict__ w, const float* __restrict__ gammas, int widx,
    __bf16* __restrict__ out, long long n) {
    float g = gammas[widx];
    for (long long i = (long long)blockIdx.x * 256 + threadIdx.x; i < n;
         i += 256LL * gridDim.x) {
        float q = rintf(w[i] / g);
        q = fminf(1.f, fmaxf(-1.f, q));
        out[i] = (__bf16)(q * g);
    }
}

__global__ __launch_bounds__(256) void f32_to_bf16(
    const float* __restrict__ x, __bf16* __restrict__ out, long long n) {
    for (long long i = (long long)blockIdx.x * 256 + threadIdx.x; i < n;
         i += 256LL * gridDim.x)
        out[i] = (__bf16)x[i];
}

// ---------------------------------------------------------------------------
// WMMA GEMM:  C[M,N] (f32) = A[M,K] (bf16, row-major) * W[N,K]^T (bf16)
// One wave per block; 32(M) x 64(N) register tile; K step 32.
// ---------------------------------------------------------------------------
__global__ __launch_bounds__(32) void gemm_bf16_f32(
    const __bf16* __restrict__ A, const __bf16* __restrict__ W,
    float* __restrict__ C, int M, int N, int K) {
    int n0 = blockIdx.x * 64;
    int m0 = blockIdx.y * 32;
    float8 acc[2][4] = {};
    for (int k0 = 0; k0 < K; k0 += 32) {
        bhalf16 a0 = load_frag(A, K, m0, k0);
        bhalf16 a1 = load_frag(A, K, m0 + 16, k0);
#pragma unroll
        for (int nt = 0; nt < 4; ++nt) {
            bhalf16 bf = load_frag(W, K, n0 + nt * 16, k0);
            acc[0][nt] = wmma_bf16(a0, bf, acc[0][nt]);
            acc[1][nt] = wmma_bf16(a1, bf, acc[1][nt]);
        }
    }
    int lane = threadIdx.x & 31;
    int colL = lane & 15;
    int rowOff = (lane & 16) ? 8 : 0;
#pragma unroll
    for (int mt = 0; mt < 2; ++mt)
#pragma unroll
        for (int nt = 0; nt < 4; ++nt)
#pragma unroll
            for (int r = 0; r < 8; ++r) {
                int row = m0 + mt * 16 + r + rowOff;
                int col = n0 + nt * 16 + colL;
                C[(size_t)row * N + col] = acc[mt][nt][r];
            }
}

// ---------------------------------------------------------------------------
// RoPE + repack Q:  qraw[M, NH*HD] f32  ->  Qb[b][h][s][d] bf16
// ---------------------------------------------------------------------------
__global__ __launch_bounds__(256) void rope_pack_q(
    const float* __restrict__ qraw, const int* __restrict__ pos_ids,
    __bf16* __restrict__ qb) {
    int bs = blockIdx.x, b = bs / S_, s = bs % S_;
    float pos = (float)pos_ids[bs];
    const float kLog = 9.2103403720f / (float)HD_;  // ln(10000)/HD
    for (int idx = threadIdx.x; idx < NH_ * (HD_ / 2); idx += blockDim.x) {
        int h = idx >> 6, j = idx & 63;
        float inv = __expf(-(float)(2 * j) * kLog);
        float fr = pos * inv, sn, cs;
        __sincosf(fr, &sn, &cs);
        float q1 = qraw[(size_t)bs * (NH_ * HD_) + h * HD_ + j];
        float q2 = qraw[(size_t)bs * (NH_ * HD_) + h * HD_ + j + 64];
        size_t o = (((size_t)b * NH_ + h) * S_ + s) * HD_;
        qb[o + j]      = (__bf16)(q1 * cs - q2 * sn);
        qb[o + j + 64] = (__bf16)(q2 * cs + q1 * sn);
    }
}

__global__ __launch_bounds__(256) void rope_pack_k(
    const float* __restrict__ kraw, const int* __restrict__ pos_ids,
    __bf16* __restrict__ kb) {
    int bs = blockIdx.x, b = bs / S_, s = bs % S_;
    float pos = (float)pos_ids[bs];
    const float kLog = 9.2103403720f / (float)HD_;
    for (int idx = threadIdx.x; idx < NKV_ * (HD_ / 2); idx += blockDim.x) {
        int kv = idx >> 6, j = idx & 63;
        float inv = __expf(-(float)(2 * j) * kLog);
        float fr = pos * inv, sn, cs;
        __sincosf(fr, &sn, &cs);
        float k1 = kraw[(size_t)bs * (NKV_ * HD_) + kv * HD_ + j];
        float k2 = kraw[(size_t)bs * (NKV_ * HD_) + kv * HD_ + j + 64];
        size_t o = (((size_t)b * NKV_ + kv) * S_ + s) * HD_;
        kb[o + j]      = (__bf16)(k1 * cs - k2 * sn);
        kb[o + j + 64] = (__bf16)(k2 * cs + k1 * sn);
    }
}

// V -> transposed bf16: Vt[b][kv][d][s]  (B-fragment columns contiguous in t)
__global__ __launch_bounds__(256) void pack_v_t(
    const float* __restrict__ vraw, __bf16* __restrict__ vt) {
    int bs = blockIdx.x, b = bs / S_, s = bs % S_;
    for (int idx = threadIdx.x; idx < NKV_ * HD_; idx += blockDim.x) {
        int kv = idx >> 7, d = idx & 127;
        vt[(((size_t)b * NKV_ + kv) * HD_ + d) * S_ + s] =
            (__bf16)vraw[(size_t)bs * (NKV_ * HD_) + idx];
    }
}

// ---------------------------------------------------------------------------
// Flash attention: one wave per 16-row q tile. Scores via WMMA (K=HD),
// online softmax in f32 fragments, P staged through LDS -> A-fragment,
// PV via WMMA on transposed V. Causal mask computed from indices.
// Ctx written as [b][s][h*HD] bf16 (the A matrix of the final GEMM).
// ---------------------------------------------------------------------------
__global__ __launch_bounds__(32) void attn_flash(
    const __bf16* __restrict__ Qb, const __bf16* __restrict__ Kb,
    const __bf16* __restrict__ Vt, __bf16* __restrict__ Ctx) {
    __shared__ __bf16 Plds[16 * 32];
    int m0 = blockIdx.x * 16;
    int h  = blockIdx.y;
    int b  = blockIdx.z;
    int kv = h / (NH_ / NKV_);
    const __bf16* Qp = Qb + (((size_t)b * NH_ + h) * S_ + m0) * HD_;
    const __bf16* Kp = Kb + ((size_t)b * NKV_ + kv) * S_ * HD_;
    const __bf16* Vp = Vt + ((size_t)b * NKV_ + kv) * HD_ * S_;

    int lane = threadIdx.x & 31;
    int colL = lane & 15;
    int rowOff = (lane & 16) ? 8 : 0;

    bhalf16 qf[4];
#pragma unroll
    for (int kk = 0; kk < 4; ++kk) qf[kk] = load_frag(Qp, HD_, 0, kk * 32);

    float8 acc[8] = {};
    float mrow[8], lrow[8];
#pragma unroll
    for (int r = 0; r < 8; ++r) { mrow[r] = -1e30f; lrow[r] = 0.f; }

    const float scale = 0.0883883476f;  // 1/sqrt(128)

    for (int t0 = 0; t0 < m0 + 16; t0 += 32) {
        float8 s0 = {}, s1 = {};
#pragma unroll
        for (int kk = 0; kk < 4; ++kk) {
            bhalf16 b0 = load_frag(Kp, HD_, t0, kk * 32);
            bhalf16 b1 = load_frag(Kp, HD_, t0 + 16, kk * 32);
            s0 = wmma_bf16(qf[kk], b0, s0);
            s1 = wmma_bf16(qf[kk], b1, s1);
        }
        float p0[8], p1[8];
#pragma unroll
        for (int r = 0; r < 8; ++r) {
            int row = m0 + r + rowOff;
            float v0 = s0[r] * scale + ((t0 + colL)      <= row ? 0.f : NEG_);
            float v1 = s1[r] * scale + ((t0 + 16 + colL) <= row ? 0.f : NEG_);
            float mx = fmaxf(v0, v1);
#pragma unroll
            for (int o = 1; o < 16; o <<= 1) mx = fmaxf(mx, __shfl_xor(mx, o, 32));
            float mnew  = fmaxf(mrow[r], mx);
            float alpha = __expf(mrow[r] - mnew);
            float e0 = __expf(v0 - mnew), e1 = __expf(v1 - mnew);
            float sum = e0 + e1;
#pragma unroll
            for (int o = 1; o < 16; o <<= 1) sum += __shfl_xor(sum, o, 32);
            lrow[r] = lrow[r] * alpha + sum;
            mrow[r] = mnew;
            p0[r] = e0; p1[r] = e1;
#pragma unroll
            for (int n = 0; n < 8; ++n) acc[n][r] *= alpha;
        }
        __syncthreads();  // WAR: previous PV reads of Plds done
#pragma unroll
        for (int r = 0; r < 8; ++r) {
            Plds[(r + rowOff) * 32 + colL]      = (__bf16)p0[r];
            Plds[(r + rowOff) * 32 + 16 + colL] = (__bf16)p1[r];
        }
        __syncthreads();  // RAW: P visible to all lanes
        bhalf16 pf = load_frag(&Plds[0], 32, 0, 0);
#pragma unroll
        for (int n = 0; n < 8; ++n) {
            bhalf16 vf = load_frag(Vp, S_, n * 16, t0);
            acc[n] = wmma_bf16(pf, vf, acc[n]);
        }
    }
#pragma unroll
    for (int r = 0; r < 8; ++r) {
        float inv = 1.f / lrow[r];
        int row = m0 + r + rowOff;
        size_t base = (((size_t)b * S_ + row) * NH_ + h) * HD_;
#pragma unroll
        for (int n = 0; n < 8; ++n)
            Ctx[base + n * 16 + colL] = (__bf16)(acc[n][r] * inv);
    }
}

// ---------------------------------------------------------------------------
extern "C" void kernel_launch(void* const* d_in, const int* in_sizes, int n_in,
                              void* d_out, int out_size, void* d_ws, size_t ws_size,
                              hipStream_t stream) {
    const float* hs  = (const float*)d_in[0];
    const float* Wq  = (const float*)d_in[1];
    const float* Wk  = (const float*)d_in[2];
    const float* Wv  = (const float*)d_in[3];
    const float* Wo  = (const float*)d_in[4];
    const int*   pos = (const int*)d_in[6];
    float* out = (float*)d_out;

    char* ws = (char*)d_ws;
    size_t off = 0;
    auto alloc = [&](size_t bytes) { char* p = ws + off; off = (off + bytes + 255) & ~(size_t)255; return p; };

    float*  partials = (float*)alloc(4 * 512 * sizeof(float));
    float*  gammas   = (float*)alloc(256);
    __bf16* WqQ = (__bf16*)alloc((size_t)NH_  * HD_ * H_ * 2);
    __bf16* WkQ = (__bf16*)alloc((size_t)NKV_ * HD_ * H_ * 2);
    __bf16* WvQ = (__bf16*)alloc((size_t)NKV_ * HD_ * H_ * 2);
    __bf16* WoQ = (__bf16*)alloc((size_t)H_ * NH_ * HD_ * 2);
    __bf16* Xb  = (__bf16*)alloc((size_t)M_ * H_ * 2);
    float*  Qraw = (float*)alloc((size_t)M_ * NH_ * HD_ * 4);  // reused as Ctx
    float*  Kraw = (float*)alloc((size_t)M_ * NKV_ * HD_ * 4);
    float*  Vraw = (float*)alloc((size_t)M_ * NKV_ * HD_ * 4);
    __bf16* Qb  = (__bf16*)alloc((size_t)B_ * NH_  * S_ * HD_ * 2);
    __bf16* Kb  = (__bf16*)alloc((size_t)B_ * NKV_ * S_ * HD_ * 2);
    __bf16* Vtb = (__bf16*)alloc((size_t)B_ * NKV_ * HD_ * S_ * 2);
    __bf16* Ctx = (__bf16*)Qraw;  // Qraw is dead after rope_pack_q

    const long long nWq = (long long)NH_ * HD_ * H_;   // 16.7M
    const long long nWk = (long long)NKV_ * HD_ * H_;  // 4.2M
    const long long nX  = (long long)M_ * H_;

    // 1) gamma = mean(|W|)+1e-5
    abs_sum_partial<<<512, 256, 0, stream>>>(Wq, nWq, partials + 0 * 512);
    abs_sum_partial<<<512, 256, 0, stream>>>(Wk, nWk, partials + 1 * 512);
    abs_sum_partial<<<512, 256, 0, stream>>>(Wv, nWk, partials + 2 * 512);
    abs_sum_partial<<<512, 256, 0, stream>>>(Wo, nWq, partials + 3 * 512);
    gamma_finalize<<<4, 256, 0, stream>>>(partials, gammas);

    // 2) quantize weights + convert activations to bf16
    quant_bf16<<<4096, 256, 0, stream>>>(Wq, gammas, 0, WqQ, nWq);
    quant_bf16<<<4096, 256, 0, stream>>>(Wk, gammas, 1, WkQ, nWk);
    quant_bf16<<<4096, 256, 0, stream>>>(Wv, gammas, 2, WvQ, nWk);
    quant_bf16<<<4096, 256, 0, stream>>>(Wo, gammas, 3, WoQ, nWq);
    f32_to_bf16<<<4096, 256, 0, stream>>>(hs, Xb, nX);

    // 3) QKV projections (WMMA bf16 -> f32)
    gemm_bf16_f32<<<dim3((NH_ * HD_) / 64, M_ / 32), 32, 0, stream>>>(Xb, WqQ, Qraw, M_, NH_ * HD_, H_);
    gemm_bf16_f32<<<dim3((NKV_ * HD_) / 64, M_ / 32), 32, 0, stream>>>(Xb, WkQ, Kraw, M_, NKV_ * HD_, H_);
    gemm_bf16_f32<<<dim3((NKV_ * HD_) / 64, M_ / 32), 32, 0, stream>>>(Xb, WvQ, Vraw, M_, NKV_ * HD_, H_);

    // 4) RoPE + repack to attention layouts
    rope_pack_q<<<B_ * S_, 256, 0, stream>>>(Qraw, pos, Qb);
    rope_pack_k<<<B_ * S_, 256, 0, stream>>>(Kraw, pos, Kb);
    pack_v_t<<<B_ * S_, 256, 0, stream>>>(Vraw, Vtb);

    // 5) flash attention (causal mask from indices)
    attn_flash<<<dim3(S_ / 16, NH_, B_), 32, 0, stream>>>(Qb, Kb, Vtb, Ctx);

    // 6) output projection -> d_out (f32)
    gemm_bf16_f32<<<dim3(H_ / 64, M_ / 32), 32, 0, stream>>>(Ctx, WoQ, out, M_, H_, NH_ * HD_);
}
